// GCN_51797305589934
// MI455X (gfx1250) — matrix operand order repaired and checked by reading, not compiled, not run on hardware
//
#include <hip/hip_runtime.h>

typedef __attribute__((ext_vector_type(16))) __bf16 v16bf;
typedef __attribute__((ext_vector_type(8)))  __bf16 v8bf;
typedef __attribute__((ext_vector_type(8)))  float  v8f;

// ---------------------------------------------------------------- utilities
__global__ void fill_f32(float* __restrict__ p, float v, long long n) {
  long long i  = (long long)blockIdx.x * blockDim.x + threadIdx.x;
  long long st = (long long)gridDim.x * blockDim.x;
  for (; i < n; i += st) p[i] = v;
}

// deg[dst] += ew  (deg pre-filled with 1.0 for the self loop)
__global__ void deg_accum(const int* __restrict__ dst, const float* __restrict__ ew,
                          float* __restrict__ deg, int E) {
  int i  = blockIdx.x * blockDim.x + threadIdx.x;
  int st = gridDim.x * blockDim.x;
  for (; i < E; i += st) unsafeAtomicAdd(&deg[dst[i]], ew[i]);
}

// dis = rsqrt(deg); deg >= 1 always (self loop)
__global__ void make_dis(float* __restrict__ deg, int n) {
  int i = blockIdx.x * blockDim.x + threadIdx.x;
  if (i < n) deg[i] = rsqrtf(deg[i]);
}

// ---- pack W[K,N] f32 -> Wp bf16 in B-fragment order: Wp[kb*N*32 + n*32 + kk],
//      k = kb*32+kk, zero-padded for k >= K. Lane fragment = 16 contiguous bf16.
__global__ void pack_w(const float* __restrict__ W, __bf16* __restrict__ Wp,
                       int K, int K2, int N) {
  long long total = (long long)K2 * N;
  long long i  = (long long)blockIdx.x * blockDim.x + threadIdx.x;
  long long st = (long long)gridDim.x * blockDim.x;
  for (; i < total; i += st) {
    int kk = (int)(i & 31);
    long long r = i >> 5;
    int n  = (int)(r % N);
    int kb = (int)(r / N);
    int k  = kb * 32 + kk;
    Wp[i] = (k < K) ? (__bf16)W[(size_t)k * N + n] : (__bf16)0.0f;
  }
}

// ---- pack X[Nn,K] f32 -> row-major bf16 [Nn,K2], K zero-padded to K2 (layer 1 only)
__global__ void pack_x(const float* __restrict__ X, __bf16* __restrict__ Xb,
                       int Nn, int K, int K2) {
  long long total = (long long)Nn * K2;
  long long i  = (long long)blockIdx.x * blockDim.x + threadIdx.x;
  long long st = (long long)gridDim.x * blockDim.x;
  for (; i < total; i += st) {
    int k   = (int)(i % K2);
    int row = (int)(i / K2);
    Xb[i] = (k < K) ? (__bf16)X[(size_t)row * K + k] : (__bf16)0.0f;
  }
}

// ------------------------------------------------- WMMA GEMM  Y = Xb(bf16) @ Wp(bf16)
// Per block: async-copy the whole packed-B tile (K2 x 16*NT cols) into LDS once
// (global_load_async_to_lds_b128, ASYNCcnt), then branch-free K loop:
// 2x16B A global loads + NT x (2x ds_load_b128) B LDS reads + NT WMMAs per K-step.
// A/B fragments use contiguous k = 16*half + s (wave32). K2 multiple of 32; M multiple of 16.
template <int NT>
__global__ void gemm_wmma_bf16(const __bf16* __restrict__ Xb, const __bf16* __restrict__ Wp,
                               float* __restrict__ Y, int M, int K2, int N) {
  __shared__ __align__(32) unsigned char bsm[16384];   // up to (128/32)*NT*1KB

  const int tid  = threadIdx.x;
  const int lane = tid & 31;
  const int wv   = tid >> 5;                       // 4 waves / block
  const int m0   = (blockIdx.x * 4 + wv) * 16;     // M tile origin
  const int n0   = blockIdx.y * (16 * NT);
  const int half = lane >> 4;
  const int l15  = lane & 15;
  const bool active = (m0 < M);                    // wave-uniform

  // ---- stage packed B tile into LDS (contiguous chunk of NT*1KB per 32-K block)
  {
    const int kbs     = K2 >> 5;
    const int chunkU  = NT * 64;                   // 16B units per kb chunk (NT*1KB)
    const int units   = kbs * chunkU;
    const unsigned char* wp0 = (const unsigned char*)Wp + (size_t)n0 * 64; // n0*32*2B
    for (int u = tid; u < units; u += blockDim.x) {
      int kb = u / chunkU;
      int r  = u - kb * chunkU;
      unsigned lds32 = (unsigned)(uintptr_t)(&bsm[(size_t)u * 16]);
      unsigned long long g64 =
          (unsigned long long)(uintptr_t)(wp0 + (size_t)kb * ((size_t)N * 64) +
                                          (size_t)r * 16);
      asm volatile("global_load_async_to_lds_b128 %0, %1, off"
                   :: "v"(lds32), "v"(g64) : "memory");
    }
    asm volatile("s_wait_asynccnt 0" ::: "memory");
  }
  __syncthreads();

  if (active) {
    v8f acc[NT] = {};

    // A: row m0+l15, lane covers K = half*16 + [0..15]
    const __bf16* xrow = Xb + (size_t)(m0 + l15) * K2 + half * 16;
    // B in LDS: lane fragment = 16 contiguous bf16 at kb*NT*512 + nl*32 + half*16
    const __bf16* bl = (const __bf16*)bsm;

    for (int k0 = 0; k0 < K2; k0 += 32) {
      v16bf a = *(const v16bf*)(xrow + k0);
      const __bf16* bk = bl + (size_t)(k0 >> 5) * (NT * 512) + half * 16;
#pragma unroll
      for (int t = 0; t < NT; ++t) {
        v16bf b = *(const v16bf*)(bk + (size_t)(t * 16 + l15) * 32);
        acc[t] = __builtin_amdgcn_wmma_f32_16x16x32_bf16(
            false, a, false, b, (short)0, acc[t], false, false);
      }
    }

    // C/D layout: VGPR r -> M = r + 8*half, N = lane&15
#pragma unroll
    for (int t = 0; t < NT; ++t)
#pragma unroll
      for (int r = 0; r < 8; ++r) {
        int row = m0 + r + half * 8;
        Y[(size_t)row * N + n0 + t * 16 + l15] = acc[t][r];
      }
  }
}

// ---------------------------------------------- edge scatter: agg[dst] += norm*xw[src]
// One wave per edge: coalesced row gather + hardware f32 atomics (L2-resident).
__global__ void scatter_edges(const float* __restrict__ xw, const int* __restrict__ src,
                              const int* __restrict__ dst, const float* __restrict__ ew,
                              const float* __restrict__ dis, float* __restrict__ agg,
                              int E, int F) {
  const int lane = threadIdx.x & 31;
  int wid = (blockIdx.x * blockDim.x + threadIdx.x) >> 5;
  int nw  = (gridDim.x * blockDim.x) >> 5;
  for (int e = wid; e < E; e += nw) {
    int s = src[e], d = dst[e];
    float norm = dis[s] * ew[e] * dis[d];
    const float* xr = xw  + (size_t)s * F;
    float*       ar = agg + (size_t)d * F;
    int e2 = e + nw;                               // prefetch next source row into L2
    if (e2 < E) __builtin_prefetch(xw + (size_t)src[e2] * F, 0, 1);
    for (int f = lane; f < F; f += 32)
      unsafeAtomicAdd(ar + f, norm * xr[f]);
  }
}

// -------- fused: v = agg + dis^2*xw (self loop) + bias; optional ReLU; column stats
// blockDim = 256, F is a power of two in {32,64,128,256}
template <int PRE_RELU>
__global__ void finalize_stats(float* __restrict__ agg, const float* __restrict__ xw,
                               const float* __restrict__ dis, const float* __restrict__ bias,
                               float* __restrict__ stats, int Nn, int F) {
  const int tid  = threadIdx.x;
  const int c    = tid & (F - 1);
  const int rpb  = 256 / F;
  const int roff = tid / F;
  float s = 0.0f, ss = 0.0f;
  float bc = bias[c];
  for (long long base = (long long)blockIdx.x * rpb; base < Nn;
       base += (long long)gridDim.x * rpb) {
    int row = (int)base + roff;
    if (row < Nn) {
      float dv  = dis[row];
      size_t ix = (size_t)row * F + c;
      float v = agg[ix] + dv * dv * xw[ix] + bc;
      if (PRE_RELU) v = fmaxf(v, 0.0f);
      agg[ix] = v;
      s += v; ss += v * v;
    }
  }
  __shared__ float sh0[256], sh1[256];
  sh0[tid] = s; sh1[tid] = ss;
  __syncthreads();
  if (tid < F) {
    float ts = 0.0f, tss = 0.0f;
    for (int j = tid; j < 256; j += F) { ts += sh0[j]; tss += sh1[j]; }
    unsafeAtomicAdd(&stats[tid], ts);
    unsafeAtomicAdd(&stats[F + tid], tss);
  }
}

// -------- BN apply (biased var, eps=1e-5), optional post-ReLU; writes f32 in place
//          AND the bf16 copy consumed by the next layer's WMMA GEMM.
template <int POST_RELU>
__global__ void bn_apply(float* __restrict__ h, __bf16* __restrict__ hbf,
                         const float* __restrict__ stats,
                         const float* __restrict__ g, const float* __restrict__ beta,
                         long long total, int F, float invN) {
  long long i  = (long long)blockIdx.x * blockDim.x + threadIdx.x;
  long long st = (long long)gridDim.x * blockDim.x;
  for (; i < total; i += st) {
    int c = (int)(i & (F - 1));
    float mean = stats[c] * invN;
    float var  = stats[F + c] * invN - mean * mean;
    float rs   = rsqrtf(var + 1e-5f);
    float sc   = g[c] * rs;
    float sf   = beta[c] - mean * sc;
    float v = h[i] * sc + sf;
    if (POST_RELU) v = fmaxf(v, 0.0f);
    h[i]   = v;
    hbf[i] = (__bf16)v;
  }
}

// -------- global_add_pool into [G, 256]
__global__ void pool_kernel(const float* __restrict__ h, const int* __restrict__ batch,
                            float* __restrict__ pooled, int Nn) {
  long long total = (long long)Nn * 256;
  long long i  = (long long)blockIdx.x * blockDim.x + threadIdx.x;
  long long st = (long long)gridDim.x * blockDim.x;
  for (; i < total; i += st) {
    int row = (int)(i >> 8);
    int c   = (int)(i & 255);
    unsafeAtomicAdd(&pooled[(size_t)batch[row] * 256 + c], h[i]);
  }
}

// -------- fc1: relu(relu(pooled) @ W + b), pooled [G,256], W [256,128]
__global__ void fc1_kernel(const float* __restrict__ pooled, const float* __restrict__ w,
                           const float* __restrict__ b, float* __restrict__ out) {
  int gi = blockIdx.x;
  int c  = threadIdx.x;
  const float* pr = pooled + (size_t)gi * 256;
  float acc = b[c];
#pragma unroll 4
  for (int k = 0; k < 256; ++k)
    acc += fmaxf(pr[k], 0.0f) * w[(size_t)k * 128 + c];
  out[(size_t)gi * 128 + c] = fmaxf(acc, 0.0f);
}

// -------- fc2: x [G,128] @ w[128] + b  -> out[G]
__global__ void fc2_kernel(const float* __restrict__ x, const float* __restrict__ w,
                           const float* __restrict__ b, float* __restrict__ out, int G) {
  int gi = blockIdx.x * blockDim.x + threadIdx.x;
  if (gi >= G) return;
  const float* xr = x + (size_t)gi * 128;
  float acc = b[0];
#pragma unroll 4
  for (int k = 0; k < 128; ++k) acc += xr[k] * w[k];
  out[gi] = acc;
}

// ---------------------------------------------------------------- launch
extern "C" void kernel_launch(void* const* d_in, const int* in_sizes, int n_in,
                              void* d_out, int out_size, void* d_ws, size_t ws_size,
                              hipStream_t stream) {
  const float* x     = (const float*)d_in[0];
  const int*   ei    = (const int*)d_in[1];
  const float* ew    = (const float*)d_in[2];
  const int*   batch = (const int*)d_in[3];
  // d_in[4] physics_score: unused by the reference
  const float* Wl[5]; const float* Bl[5]; const float* Gl[5]; const float* Be[5];
  for (int l = 0; l < 5; ++l) {
    Wl[l] = (const float*)d_in[5 + 4 * l + 0];
    Bl[l] = (const float*)d_in[5 + 4 * l + 1];
    Gl[l] = (const float*)d_in[5 + 4 * l + 2];
    Be[l] = (const float*)d_in[5 + 4 * l + 3];
  }
  const float* fc1w = (const float*)d_in[25];
  const float* fc1b = (const float*)d_in[26];
  const float* fc2w = (const float*)d_in[27];
  const float* fc2b = (const float*)d_in[28];

  const int Nn = in_sizes[3];          // 50000 nodes
  const int E  = in_sizes[1] / 2;      // 800000 edges
  const int G  = in_sizes[4];          // 512 graphs
  const int Fs[6]  = {20, 32, 64, 128, 128, 256};
  const int K2s[5] = {32, 32, 64, 128, 128};   // K padded to multiple of 32
  const int FMAX = 256;

  // ---- workspace carve
  char* p = (char*)d_ws;
  auto alloc = [&](size_t bytes) -> void* {
    void* r = (void*)p;
    p += (bytes + 255) & ~(size_t)255;
    return r;
  };
  float*  buf0   = (float*)alloc((size_t)Nn * FMAX * 4);  // h / agg (f32)
  float*  buf1   = (float*)alloc((size_t)Nn * FMAX * 4);  // xw      (f32)
  __bf16* hbf    = (__bf16*)alloc((size_t)Nn * FMAX * 2); // GEMM A operand (bf16)
  __bf16* wpack  = (__bf16*)alloc((size_t)128 * FMAX * 2);// packed W (bf16), max 64KB
  float*  dis    = (float*)alloc((size_t)Nn * 4);
  float*  stats  = (float*)alloc((size_t)2 * FMAX * 4);
  float*  pooled = (float*)alloc((size_t)G * FMAX * 4);
  float*  fc1o   = (float*)alloc((size_t)G * 128 * 4);

  const int* src = ei;
  const int* dst = ei + E;

  // ---- degree -> dis = rsqrt(1 + sum_in ew)
  fill_f32<<<256, 256, 0, stream>>>(dis, 1.0f, Nn);
  deg_accum<<<1024, 256, 0, stream>>>(dst, ew, dis, E);
  make_dis<<<(Nn + 255) / 256, 256, 0, stream>>>(dis, Nn);

  // ---- layer-1 A operand: x [Nn,20] -> bf16 [Nn,32] zero-padded
  pack_x<<<1024, 256, 0, stream>>>(x, hbf, Nn, Fs[0], K2s[0]);

  const float invN = 1.0f / (float)Nn;
  const int mblocks = ((Nn + 15) / 16 + 3) / 4;   // 4 M-tiles (waves) per block

  for (int l = 0; l < 5; ++l) {
    const int K = Fs[l], K2 = K2s[l], F = Fs[l + 1];

    // 0) pack weights into B-fragment order
    pack_w<<<64, 256, 0, stream>>>(Wl[l], wpack, K, K2, F);

    // 1) xw = h @ W  (WMMA bf16 -> f32); reads hbf written by pack_x / previous bn
    if (F >= 64) {
      dim3 grid(mblocks, F / 64);
      gemm_wmma_bf16<4><<<grid, 128, 0, stream>>>(hbf, wpack, buf1, Nn, K2, F);
    } else {
      dim3 grid(mblocks, F / 32);
      gemm_wmma_bf16<2><<<grid, 128, 0, stream>>>(hbf, wpack, buf1, Nn, K2, F);
    }

    // 2) zero agg, scatter edges
    fill_f32<<<2048, 256, 0, stream>>>(buf0, 0.0f, (long long)Nn * F);
    scatter_edges<<<2048, 256, 0, stream>>>(buf1, src, dst, ew, dis, buf0, E, F);

    // 3) fused self-loop + bias + (pre-)ReLU + column stats
    fill_f32<<<1, 256, 0, stream>>>(stats, 0.0f, 2 * F);
    if (l < 3)
      finalize_stats<1><<<512, 256, 0, stream>>>(buf0, buf1, dis, Bl[l], stats, Nn, F);
    else
      finalize_stats<0><<<512, 256, 0, stream>>>(buf0, buf1, dis, Bl[l], stats, Nn, F);

    // 4) BN apply (+ post-ReLU on layer 4); emits f32 (for pooling) + bf16 (next GEMM)
    long long total = (long long)Nn * F;
    if (l == 3)
      bn_apply<1><<<2048, 256, 0, stream>>>(buf0, hbf, stats, Gl[l], Be[l], total, F, invN);
    else
      bn_apply<0><<<2048, 256, 0, stream>>>(buf0, hbf, stats, Gl[l], Be[l], total, F, invN);
  }

  // ---- pool + MLP head
  fill_f32<<<256, 256, 0, stream>>>(pooled, 0.0f, (long long)G * 256);
  pool_kernel<<<2048, 256, 0, stream>>>(buf0, batch, pooled, Nn);
  fc1_kernel<<<G, 128, 0, stream>>>(pooled, fc1w, fc1b, fc1o);
  fc2_kernel<<<(G + 255) / 256, 256, 0, stream>>>(fc1o, fc2w, fc2b, (float*)d_out, G);
}
